// CryoProjector_21492016349235
// MI455X (gfx1250) — matrix operand order repaired and checked by reading, not compiled, not run on hardware
//
#include <hip/hip_runtime.h>
#include <math.h>

typedef __attribute__((ext_vector_type(16))) _Float16 v16h;
typedef __attribute__((ext_vector_type(8)))  float    v8f;

#define BB 4
#define NN 2048          // atoms (K dimension)
#define HH 128
#define WW 128
#define KC (NN / 32)     // 64 K-chunks of 32 atoms
#define TM (HH / 16)     // 8 row tiles
#define TW (WW / 16)     // 8 col tiles
#define FRAG_HALFS 512   // one 16x32 (or 32x16) f16 fragment = 512 halfs

// -log2(e) / (2 * sigma^2), sigma = 1.5  ->  exp(-d^2/4.5) = exp2(C2 * d^2)
#define C2 (-0.32056992f)

// ---------------------------------------------------------------------------
// Kernel 1: rotate atoms, compute projected pixel coords px/py; copy rot out.
// ---------------------------------------------------------------------------
__global__ __launch_bounds__(256) void proj_kernel(
    const float* __restrict__ x, const float* __restrict__ rot,
    float* __restrict__ px, float* __restrict__ py,
    float* __restrict__ rot_out) {
  int tid = blockIdx.x * blockDim.x + threadIdx.x;
  if (tid < BB * 9) rot_out[tid] = rot[tid];
  if (tid >= BB * NN) return;
  int b = tid / NN;
  const float* xp = x + (size_t)tid * 3;
  float x0 = xp[0], x1 = xp[1], x2 = xp[2];
  const float* r = rot + b * 9;
  const float scale = 0.8f * 0.5f * (float)(HH < WW ? HH : WW);  // 51.2
  float xr0 = r[0] * x0 + r[1] * x1 + r[2] * x2;
  float xr1 = r[3] * x0 + r[4] * x1 + r[5] * x2;
  px[tid] = xr0 * scale + (float)WW * 0.5f;
  py[tid] = xr1 * scale + (float)HH * 0.5f;
}

// ---------------------------------------------------------------------------
// Kernel 2: build f16 Gaussian factor fragments directly in WMMA register
// layout (ISA 7.12.2), so the GEMM loads are contiguous 32B per lane.
//   side 0 -> Ey as A-matrix (16x32, M=row h, K=atom)
//   side 1 -> Ex as B-matrix (32x16, K=atom, N=col w)
// ---------------------------------------------------------------------------
__global__ __launch_bounds__(256) void frag_kernel(
    const float* __restrict__ px, const float* __restrict__ py,
    _Float16* __restrict__ exfrag, _Float16* __restrict__ eyfrag) {
  int tid = blockIdx.x * blockDim.x + threadIdx.x;
  const int per_side = BB * TM * KC * 32;
  int side = (tid >= per_side) ? 1 : 0;
  int t = side ? tid - per_side : tid;
  int lane = t & 31;  t >>= 5;
  int kc   = t % KC;  t /= KC;
  int tile = t % TM;  t /= TM;
  int b = t;
  const int atom_base = b * NN + kc * 32;
  v16h frag;
  if (side == 0) {
    // A-fragment (Ey): lane row = lane%16; K(j,lane) per 16-bit A 16x32 table
    float gy = (float)(tile * 16 + (lane & 15));
#pragma unroll
    for (int j = 0; j < 16; ++j) {
      int kl = (j & 7) + ((j >> 3) << 4) + ((lane >> 4) << 3);
      float d = gy - py[atom_base + kl];
      frag[j] = (_Float16)exp2f(C2 * d * d);
    }
    *(v16h*)(eyfrag + ((size_t)((b * TM + tile) * KC + kc)) * FRAG_HALFS +
             lane * 16) = frag;
  } else {
    // B-fragment (Ex): lane col = lane%16; K(j,lane) same table order as A
    float gx = (float)(tile * 16 + (lane & 15));
#pragma unroll
    for (int j = 0; j < 16; ++j) {
      int kl = (j & 7) + ((j >> 3) << 4) + ((lane >> 4) << 3);
      float d = gx - px[atom_base + kl];
      frag[j] = (_Float16)exp2f(C2 * d * d);
    }
    *(v16h*)(exfrag + ((size_t)((b * TW + tile) * KC + kc)) * FRAG_HALFS +
             lane * 16) = frag;
  }
}

// ---------------------------------------------------------------------------
// Kernel 3: batched GEMM clean[b] = Ey[b] (128xK) x Ex[b]^T (Kx128) using
// v_wmma_f32_16x16x32_f16. One wave owns a 32x32 output tile (2x2 fragment
// grid -> each A/B fragment feeds two WMMAs). Fused epilogue writes clean and
// noisy = clean + 0.1*noise straight from the C/D register layout.
// ---------------------------------------------------------------------------
__device__ __forceinline__ void store_tile(
    const v8f& acc, int b, int row0, int col, int lane,
    const float* __restrict__ noise, float* __restrict__ noisy,
    float* __restrict__ clean) {
  int rbase = row0 + ((lane >> 4) << 3);  // +8 rows for lanes 16..31
#pragma unroll
  for (int r = 0; r < 8; ++r) {
    int idx = (b * HH + rbase + r) * WW + col;
    float v = acc[r];
    clean[idx] = v;
    noisy[idx] = v + 0.1f * noise[idx];
  }
}

__global__ __launch_bounds__(128) void gemm_kernel(
    const _Float16* __restrict__ exfrag, const _Float16* __restrict__ eyfrag,
    const float* __restrict__ noise, float* __restrict__ noisy,
    float* __restrict__ clean) {
  const int lane = threadIdx.x & 31;
  const int wave = threadIdx.x >> 5;
  int t = blockIdx.x * 4 + wave;          // 64 tiles total
  int b   = t >> 4;
  int rem = t & 15;
  int tm2 = rem >> 2;                     // 32-row group (0..3)
  int tw2 = rem & 3;                      // 32-col group (0..3)

  const _Float16* a0p = eyfrag +
      ((size_t)((b * TM + tm2 * 2 + 0) * KC)) * FRAG_HALFS + lane * 16;
  const _Float16* a1p = eyfrag +
      ((size_t)((b * TM + tm2 * 2 + 1) * KC)) * FRAG_HALFS + lane * 16;
  const _Float16* b0p = exfrag +
      ((size_t)((b * TW + tw2 * 2 + 0) * KC)) * FRAG_HALFS + lane * 16;
  const _Float16* b1p = exfrag +
      ((size_t)((b * TW + tw2 * 2 + 1) * KC)) * FRAG_HALFS + lane * 16;

  v8f acc00 = {}, acc01 = {}, acc10 = {}, acc11 = {};
  for (int kc = 0; kc < KC; ++kc) {
    v16h a0 = *(const v16h*)(a0p + (size_t)kc * FRAG_HALFS);
    v16h a1 = *(const v16h*)(a1p + (size_t)kc * FRAG_HALFS);
    v16h bm0 = *(const v16h*)(b0p + (size_t)kc * FRAG_HALFS);
    v16h bm1 = *(const v16h*)(b1p + (size_t)kc * FRAG_HALFS);
    acc00 = __builtin_amdgcn_wmma_f32_16x16x32_f16(false, a0, false, bm0,
                                                   (short)0, acc00, false, false);
    acc01 = __builtin_amdgcn_wmma_f32_16x16x32_f16(false, a0, false, bm1,
                                                   (short)0, acc01, false, false);
    acc10 = __builtin_amdgcn_wmma_f32_16x16x32_f16(false, a1, false, bm0,
                                                   (short)0, acc10, false, false);
    acc11 = __builtin_amdgcn_wmma_f32_16x16x32_f16(false, a1, false, bm1,
                                                   (short)0, acc11, false, false);
  }

  int row0 = tm2 * 32;
  int col0 = tw2 * 32 + (lane & 15);
  store_tile(acc00, b, row0 +  0, col0 +  0, lane, noise, noisy, clean);
  store_tile(acc01, b, row0 +  0, col0 + 16, lane, noise, noisy, clean);
  store_tile(acc10, b, row0 + 16, col0 +  0, lane, noise, noisy, clean);
  store_tile(acc11, b, row0 + 16, col0 + 16, lane, noise, noisy, clean);
}

// ---------------------------------------------------------------------------
extern "C" void kernel_launch(void* const* d_in, const int* in_sizes, int n_in,
                              void* d_out, int out_size, void* d_ws,
                              size_t ws_size, hipStream_t stream) {
  const float* x     = (const float*)d_in[0];  // (B,N,3)
  const float* rot   = (const float*)d_in[1];  // (B,3,3)
  const float* noise = (const float*)d_in[2];  // (B,H,W)

  // outputs concatenated: noisy (B*H*W), rot (B*9), clean (B*H*W)
  float* noisy   = (float*)d_out;
  float* rot_out = noisy + (size_t)BB * HH * WW;
  float* clean   = rot_out + BB * 9;

  // workspace carve-up (~4.3 MB)
  char* ws = (char*)d_ws;
  float* px = (float*)ws;
  float* py = px + BB * NN;
  _Float16* eyfrag = (_Float16*)(ws + (size_t)2 * BB * NN * sizeof(float));
  _Float16* exfrag = eyfrag + (size_t)BB * TM * KC * FRAG_HALFS;

  proj_kernel<<<(BB * NN + 255) / 256, 256, 0, stream>>>(x, rot, px, py,
                                                         rot_out);
  int frag_threads = 2 * BB * TM * KC * 32;  // 131072
  frag_kernel<<<frag_threads / 256, 256, 0, stream>>>(px, py, exfrag, eyfrag);
  gemm_kernel<<<16, 128, 0, stream>>>(exfrag, eyfrag, noise, noisy, clean);
}
